// SingleStreamBlockLoraProcessor_64811056496724
// MI455X (gfx1250) — compile-verified, matrix operand order checked by reading, not run on hardware
//
#include <hip/hip_runtime.h>
#include <hip/hip_bf16.h>
#include <stdint.h>

#define SEQ 2560
#define DIM 3072
#define HEADS 24
#define HD 128
#define RANK 16
#define D3 9216            // 3*DIM
#define H1 21504           // 3*DIM + MLP
#define CATD 15360         // DIM + MLP
#define EPSV 1e-6f
#define ASTRIDE 40         // padded LDS row stride (halfs) for 32-wide K tiles

typedef unsigned short u16;
typedef __attribute__((ext_vector_type(16))) __bf16 v16bf;
typedef __attribute__((ext_vector_type(8)))  float  v8f;
typedef __attribute__((ext_vector_type(4)))  unsigned int u32x4;
typedef __attribute__((ext_vector_type(8)))  int i32x8;
typedef __attribute__((ext_vector_type(4)))  int i32x4;

// Tensor Data Mover path (guarded: falls back to manual staging if absent)
#if defined(__has_builtin)
#if __has_builtin(__builtin_amdgcn_tensor_load_to_lds) && __has_builtin(__builtin_amdgcn_s_wait_tensorcnt)
#define USE_TDM 1
#endif
#endif
#ifndef USE_TDM
#define USE_TDM 0
#endif

__device__ __forceinline__ v8f zero8() {
  v8f z = {0.f, 0.f, 0.f, 0.f, 0.f, 0.f, 0.f, 0.f};
  return z;
}
__device__ __forceinline__ u16 f2bf(float f) {
  unsigned u = __float_as_uint(f);
  u += 0x7FFFu + ((u >> 16) & 1u);
  return (u16)(u >> 16);
}
__device__ __forceinline__ float bf2f(u16 b) {
  return __uint_as_float(((unsigned)b) << 16);
}
// tanh-approx GELU in sigmoid form: branch-free, one v_exp_f32, saturates right
__device__ __forceinline__ float gelu_fast(float x) {
  float u = 1.5957691216f * (x + 0.044715f * x * x * x);   // 2*sqrt(2/pi)*(...)
  return x / (1.f + __expf(-u));
}

union Frag { uint4 u[2]; v16bf v; };

// A/B fragment loader for v_wmma_f32_16x16x32_bf16 (wave32 layout):
// lane L holds row (L&15); elements 0-7 -> K = kb..kb+7, 8-15 -> K = kb+16..kb+23,
// kb = 8*(L>=16).  Works for both A (row=M) and B (row=N, since B[k][n]=W[n][k]).
__device__ __forceinline__ v16bf load_frag(const u16* base, int stride, int lane) {
  const u16* p = base + (lane & 15) * stride + ((lane >> 4) << 3);
  Frag f;
  f.u[0] = *(const uint4*)(p);
  f.u[1] = *(const uint4*)(p + 16);
  return f.v;
}

__device__ __forceinline__ v8f wmma_bf16(v16bf a, v16bf b, v8f c) {
  return __builtin_amdgcn_wmma_f32_16x16x32_bf16(false, a, false, b, (short)0, c, false, false);
}

#if USE_TDM
// One TDM 2D tile load: rows x rowlen bf16 elems, global row stride
// `stride_elems`, into LDS at byte offset lds_addr, LDS rows padded 64B->80B
// (pad_interval code 3 = 16 DWORDs, pad_amount code 3 = 4 DWORDs) to match
// ASTRIDE.  D# layout per CDNA5 ISA ch.8.
__device__ __forceinline__ void tdm_load_tile(const u16* gsrc, unsigned lds_addr,
                                              unsigned rows, unsigned rowlen,
                                              unsigned stride_elems) {
  unsigned long long ga = (unsigned long long)(size_t)gsrc;
  u32x4 g0;
  g0.x = 1u;                                   // count=1 (valid descriptor)
  g0.y = lds_addr;                             // [63:32] LDS byte address
  g0.z = (unsigned)(ga & 0xFFFFFFFFu);         // global addr lo
  g0.w = (unsigned)((ga >> 32) & 0x01FFFFFFu) | (2u << 30);  // addr[56:32] | type=2
  unsigned td0 = stride_elems;                 // tensor_dim0 >= rowlen (no OOB)
  unsigned td1 = 0x7FFFFFFFu;                  // tensor_dim1: large
  i32x8 g1;
  g1[0] = (int)((1u << 16) | (1u << 20) | (3u << 22) | (3u << 25)); // data_size=2B, pad on
  g1[1] = (int)((td0 & 0xFFFFu) << 16);                        // tensor_dim0 lo
  g1[2] = (int)((td0 >> 16) | ((td1 & 0xFFFFu) << 16));        // dim0 hi | dim1 lo
  g1[3] = (int)((td1 >> 16) | (rowlen << 16));                 // dim1 hi | tile_dim0
  g1[4] = (int)(rows & 0xFFFFu);                               // tile_dim1 (tile_dim2=0)
  g1[5] = (int)stride_elems;                                   // tensor_dim0_stride lo32
  g1[6] = 0;                                                   // stride hi16 | dim1_stride lo
  g1[7] = 0;
  i32x4 z4 = {0, 0, 0, 0};
#if __clang_major__ >= 23
  i32x8 z8 = {0, 0, 0, 0, 0, 0, 0, 0};
  __builtin_amdgcn_tensor_load_to_lds(g0, g1, z4, z4, z8, 0);
#else
  __builtin_amdgcn_tensor_load_to_lds(g0, g1, z4, z4, 0);
#endif
}
#endif

// ---------------- kernel 1: mod = silu(vec) @ mod_w.T + mod_b ----------------
__global__ void k_mod(const float* __restrict__ vec, const float* __restrict__ mod_w,
                      const float* __restrict__ mod_b, float* __restrict__ mod) {
  int lane = threadIdx.x & 31;
  int j = blockIdx.x * 8 + (threadIdx.x >> 5);
  const float* wr = mod_w + (size_t)j * DIM;
  float acc = 0.f;
  for (int d = lane; d < DIM; d += 32) {
    float v = vec[d];
    float s = v / (1.f + __expf(-v));
    acc += s * wr[d];
  }
#pragma unroll
  for (int m = 16; m >= 1; m >>= 1) acc += __shfl_xor(acc, m, 32);
  if (lane == 0) mod[j] = acc + mod_b[j];
}

// -------- kernel 2: layernorm + (1+scale)*xhat + shift, write bf16 -----------
__global__ void k_lnmod(const float* __restrict__ x, const float* __restrict__ mod,
                        u16* __restrict__ xmod) {
  int m = blockIdx.x;
  int tid = threadIdx.x;
  const float* xr = x + (size_t)m * DIM;
  float s = 0.f, s2 = 0.f;
  for (int d = tid; d < DIM; d += 256) { float v = xr[d]; s += v; s2 += v * v; }
#pragma unroll
  for (int w = 16; w >= 1; w >>= 1) { s += __shfl_xor(s, w, 32); s2 += __shfl_xor(s2, w, 32); }
  __shared__ float rs[8], rq[8];
  if ((tid & 31) == 0) { rs[tid >> 5] = s; rq[tid >> 5] = s2; }
  __syncthreads();
  float ts = 0.f, tq = 0.f;
#pragma unroll
  for (int i = 0; i < 8; ++i) { ts += rs[i]; tq += rq[i]; }
  float mu = ts * (1.f / DIM);
  float var = tq * (1.f / DIM) - mu * mu;
  float inv = rsqrtf(var + EPSV);
  u16* orow = xmod + (size_t)m * DIM;
  for (int d = tid; d < DIM; d += 256) {
    float v = (1.f + mod[DIM + d]) * ((xr[d] - mu) * inv) + mod[d];
    orow[d] = f2bf(v);
  }
}

// ------------ kernel 3/7: LoRA down  t[m][r] = A[m][:] . down[r][:] ----------
__global__ void k_lora_down(const u16* __restrict__ a, const float* __restrict__ down,
                            float* __restrict__ t, int K) {
  int lane = threadIdx.x & 31;
  int m = blockIdx.x * 8 + (threadIdx.x >> 5);
  const u16* ar = a + (size_t)m * K;
  float acc[RANK];
#pragma unroll
  for (int r = 0; r < RANK; ++r) acc[r] = 0.f;
  for (int d = lane; d < K; d += 32) {
    float xv = bf2f(ar[d]);
#pragma unroll
    for (int r = 0; r < RANK; ++r) acc[r] += xv * down[(size_t)r * K + d];
  }
#pragma unroll
  for (int r = 0; r < RANK; ++r) {
    float v = acc[r];
#pragma unroll
    for (int w = 16; w >= 1; w >>= 1) v += __shfl_xor(v, w, 32);
    if (lane == 0) t[m * RANK + r] = v;
  }
}

// ------------------- shared WMMA GEMM core: C = A(bf16) @ W(f32).T -----------
// block tile 128x256, 8 waves as 2(M)x4(N), wave tile 64x64, K-step 32,
// double-buffered LDS, A staged via TDM (or manual), B converted f32->bf16.
__device__ __forceinline__ void gemm_compute(const u16* As, const u16* Bs,
                                             int wm, int wn, int lane, v8f (*c)[4]) {
  v16bf af[4];
#pragma unroll
  for (int i = 0; i < 4; ++i) af[i] = load_frag(As + (size_t)(wm + i * 16) * ASTRIDE, ASTRIDE, lane);
#pragma unroll
  for (int j = 0; j < 4; ++j) {
    v16bf bfv = load_frag(Bs + (size_t)(wn + j * 16) * ASTRIDE, ASTRIDE, lane);
#pragma unroll
    for (int i = 0; i < 4; ++i) c[i][j] = wmma_bf16(af[i], bfv, c[i][j]);
  }
}

__device__ __forceinline__ void gemm_mainloop(const u16* __restrict__ A,
                                              const float* __restrict__ W, int K,
                                              int m0, int n0, u16* As, u16* Bs,
                                              v8f (*c)[4]) {
  int tid = threadIdx.x;
  int lane = tid & 31;
  int wave = tid >> 5;
  int wm = (wave & 1) * 64;
  int wn = (wave >> 1) * 64;
  int arow = tid >> 1, aseg = (tid & 1) * 16;
  int brow = tid >> 3, bcol = (tid & 7) * 4;
  const u16* aptr = A + (size_t)(m0 + arow) * K + aseg;
  const float* bptr = W + (size_t)(n0 + brow) * K + bcol;
  const int nk = K / 32;
  (void)aptr;

  auto stageB = [&](int buf, int k0) {
    u16* bb = Bs + (size_t)buf * 256 * ASTRIDE;
#pragma unroll
    for (int p = 0; p < 8; ++p) {
      float4 bv = *(const float4*)(bptr + (size_t)(p * 32) * K + k0);
      ushort4 o = {f2bf(bv.x), f2bf(bv.y), f2bf(bv.z), f2bf(bv.w)};
      *(ushort4*)(&bb[(size_t)(brow + p * 32) * ASTRIDE + bcol]) = o;
    }
    __builtin_prefetch(bptr + k0 + 128, 0, 1);
  };
  auto stageA = [&](int buf, int k0) {
#if USE_TDM
    if (wave == 0) {
      unsigned lds = (unsigned)(size_t)(const void*)&As[(size_t)buf * 128 * ASTRIDE];
      tdm_load_tile(A + (size_t)m0 * K + k0, lds, 128u, 32u, (unsigned)K);
    }
#else
    uint4 a0 = *(const uint4*)(aptr + k0);
    uint4 a1 = *(const uint4*)(aptr + k0 + 8);
    u16* ab = &As[(size_t)buf * 128 * ASTRIDE + (size_t)arow * ASTRIDE + aseg];
    *(uint4*)(ab) = a0;
    *(uint4*)(ab + 8) = a1;
#endif
  };

  stageA(0, 0);
  stageB(0, 0);
#if USE_TDM
  if (wave == 0) __builtin_amdgcn_s_wait_tensorcnt(0);
#endif
  __syncthreads();
  for (int it = 0; it < nk; ++it) {
    int buf = it & 1;
    if (it + 1 < nk) {
      stageA(buf ^ 1, (it + 1) * 32);   // writes buf^1: last read before prev barrier
      stageB(buf ^ 1, (it + 1) * 32);
    }
    gemm_compute(&As[(size_t)buf * 128 * ASTRIDE], &Bs[(size_t)buf * 256 * ASTRIDE],
                 wm, wn, lane, c);
#if USE_TDM
    if (it + 1 < nk && wave == 0) __builtin_amdgcn_s_wait_tensorcnt(0);
#endif
    __syncthreads();
  }
}

// ------- kernel 4: GEMM1  h = xmod @ lin1_w.T + b (+LoRA up / GELU split) ----
// grid: (M blocks fastest, N blocks outer) so consecutive blocks reuse the same
// weight tile out of L2 (lin1_w fp32 = 264MB > 192MB L2).
__global__ void __launch_bounds__(256)
k_gemm1(const u16* __restrict__ A, const float* __restrict__ W,
        const float* __restrict__ bias, const float* __restrict__ t1,
        const float* __restrict__ up, float* __restrict__ qkv, u16* __restrict__ cat) {
  __shared__ u16 As[2 * 128 * ASTRIDE];
  __shared__ u16 Bs[2 * 256 * ASTRIDE];
  int lane = threadIdx.x & 31;
  int wave = threadIdx.x >> 5;
  int wm = (wave & 1) * 64;
  int wn = (wave >> 1) * 64;
  int m0 = blockIdx.x * 128;
  int n0 = blockIdx.y * 256;
  v8f c[4][4];
#pragma unroll
  for (int i = 0; i < 4; ++i)
#pragma unroll
    for (int j = 0; j < 4; ++j) c[i][j] = zero8();

  gemm_mainloop(A, W, DIM, m0, n0, As, Bs, c);

  int mB = m0 + wm + ((lane >> 4) << 3);
  int nB = n0 + wn + (lane & 15);
#pragma unroll
  for (int i = 0; i < 4; ++i) {
#pragma unroll
    for (int j = 0; j < 4; ++j) {
#pragma unroll
      for (int v = 0; v < 8; ++v) {
        int m = mB + i * 16 + v;
        int n = nB + j * 16;
        float val = c[i][j][v] + bias[n];
        if (n < D3) {
          const float* upr = up + (size_t)n * RANK;
          const float* tr = t1 + (size_t)m * RANK;
          float lo = 0.f;
#pragma unroll
          for (int r = 0; r < RANK; ++r) lo += tr[r] * upr[r];
          qkv[(size_t)m * D3 + n] = val + lo;
        } else {
          cat[(size_t)m * CATD + DIM + (n - D3)] = f2bf(gelu_fast(val));
        }
      }
    }
  }
}

// ---- kernel 5: per-(pos,head) RMS norm q/k + RoPE + split to q/k/v bf16 -----
__global__ void k_qkv_prep(const float* __restrict__ qkv, const float* __restrict__ pe,
                           const float* __restrict__ qsc, const float* __restrict__ ksc,
                           u16* __restrict__ qb, u16* __restrict__ kb, u16* __restrict__ vb) {
  int lane = threadIdx.x & 31;
  int idx = blockIdx.x * 8 + (threadIdx.x >> 5);
  int m = idx / HEADS;
  int h = idx % HEADS;
  int d0 = lane * 4;
  const float* base = qkv + (size_t)m * D3 + h * HD;
  float4 q = *(const float4*)(base + d0);
  float4 k = *(const float4*)(base + DIM + d0);
  float4 v = *(const float4*)(base + 2 * DIM + d0);
  float ssq = q.x * q.x + q.y * q.y + q.z * q.z + q.w * q.w;
  float ssk = k.x * k.x + k.y * k.y + k.z * k.z + k.w * k.w;
#pragma unroll
  for (int w = 16; w >= 1; w >>= 1) { ssq += __shfl_xor(ssq, w, 32); ssk += __shfl_xor(ssk, w, 32); }
  float rq = rsqrtf(ssq * (1.f / HD) + EPSV);
  float rk = rsqrtf(ssk * (1.f / HD) + EPSV);
  float4 s4 = *(const float4*)(qsc + d0);
  float4 t4 = *(const float4*)(ksc + d0);
  q.x *= rq * s4.x; q.y *= rq * s4.y; q.z *= rq * s4.z; q.w *= rq * s4.w;
  k.x *= rk * t4.x; k.y *= rk * t4.y; k.z *= rk * t4.z; k.w *= rk * t4.w;
  const float* p0 = pe + (size_t)m * 256 + (size_t)(lane * 2) * 4;
  float c0 = p0[0], s0 = p0[2], c1 = p0[4], s1 = p0[6];
  const float iq = 0.08838834764831845f;  // 1/sqrt(HD) folded into q
  float q0 = (c0 * q.x - s0 * q.y) * iq;
  float q1 = (s0 * q.x + c0 * q.y) * iq;
  float q2 = (c1 * q.z - s1 * q.w) * iq;
  float q3 = (s1 * q.z + c1 * q.w) * iq;
  float k0 = c0 * k.x - s0 * k.y;
  float k1 = s0 * k.x + c0 * k.y;
  float k2 = c1 * k.z - s1 * k.w;
  float k3 = s1 * k.z + c1 * k.w;
  size_t o = ((size_t)h * SEQ + m) * HD + d0;
  ushort4 qo = {f2bf(q0), f2bf(q1), f2bf(q2), f2bf(q3)};
  ushort4 ko = {f2bf(k0), f2bf(k1), f2bf(k2), f2bf(k3)};
  ushort4 vo = {f2bf(v.x), f2bf(v.y), f2bf(v.z), f2bf(v.w)};
  *(ushort4*)(qb + o) = qo;
  *(ushort4*)(kb + o) = ko;
  *(ushort4*)(vb + o) = vo;
}

// ---------------- kernel 6: flash attention, WMMA bf16 -----------------------
__global__ void __launch_bounds__(256)
k_attn(const u16* __restrict__ qg, const u16* __restrict__ kg, const u16* __restrict__ vg,
       u16* __restrict__ cat) {
  __shared__ u16 kLDS[32 * 128];       // [kpos][hd]
  __shared__ u16 vLDS[128 * 40];       // [hd][kpos] (transposed, padded)
  __shared__ u16 pLDS[8 * 16 * 40];    // per-wave P scratch 16x32 (padded)
  int tid = threadIdx.x;
  int lane = tid & 31;
  int wave = tid >> 5;
  int head = blockIdx.y;
  int m0 = blockIdx.x * 128 + wave * 16;
  const u16* qbase = qg + ((size_t)head * SEQ + m0) * HD;
  v16bf qf[4];
#pragma unroll
  for (int t = 0; t < 4; ++t) qf[t] = load_frag(qbase + t * 32, HD, lane);

  float mI[8], lI[8];
  v8f O[8];
#pragma unroll
  for (int v = 0; v < 8; ++v) { mI[v] = -1e30f; lI[v] = 0.f; }
#pragma unroll
  for (int c = 0; c < 8; ++c) O[c] = zero8();

  int lpos = tid >> 3;
  int lc0 = (tid & 7) * 16;
  const u16* kbase = kg + (size_t)head * SEQ * HD;
  const u16* vbase = vg + (size_t)head * SEQ * HD;
  u16* pw = pLDS + wave * (16 * 40);

  for (int j0 = 0; j0 < SEQ; j0 += 32) {
    __syncthreads();
    const u16* ksrc = kbase + (size_t)(j0 + lpos) * HD + lc0;
    *(uint4*)(&kLDS[lpos * 128 + lc0]) = *(const uint4*)(ksrc);
    *(uint4*)(&kLDS[lpos * 128 + lc0 + 8]) = *(const uint4*)(ksrc + 8);
    const u16* vsrc = vbase + (size_t)(j0 + lpos) * HD + lc0;
#pragma unroll
    for (int i = 0; i < 16; ++i) vLDS[(lc0 + i) * 40 + lpos] = vsrc[i];
    __syncthreads();

    v8f s0 = zero8(), s1 = zero8();
#pragma unroll
    for (int t = 0; t < 4; ++t) {
      v16bf f0 = load_frag(&kLDS[t * 32], 128, lane);
      v16bf f1 = load_frag(&kLDS[16 * 128 + t * 32], 128, lane);
      s0 = wmma_bf16(qf[t], f0, s0);
      s1 = wmma_bf16(qf[t], f1, s1);
    }
    int rbase = (lane >> 4) << 3;
#pragma unroll
    for (int v = 0; v < 8; ++v) {
      float mx = fmaxf(s0[v], s1[v]);
      mx = fmaxf(mx, __shfl_xor(mx, 1, 32));
      mx = fmaxf(mx, __shfl_xor(mx, 2, 32));
      mx = fmaxf(mx, __shfl_xor(mx, 4, 32));
      mx = fmaxf(mx, __shfl_xor(mx, 8, 32));
      float mNew = fmaxf(mI[v], mx);
      float al = __expf(mI[v] - mNew);
      float p0 = __expf(s0[v] - mNew);
      float p1 = __expf(s1[v] - mNew);
      float rsum = p0 + p1;
      rsum += __shfl_xor(rsum, 1, 32);
      rsum += __shfl_xor(rsum, 2, 32);
      rsum += __shfl_xor(rsum, 4, 32);
      rsum += __shfl_xor(rsum, 8, 32);
      lI[v] = lI[v] * al + rsum;
      mI[v] = mNew;
      int r = v + rbase;
      pw[r * 40 + (lane & 15)] = f2bf(p0);
      pw[r * 40 + 16 + (lane & 15)] = f2bf(p1);
#pragma unroll
      for (int c = 0; c < 8; ++c) O[c][v] *= al;
    }
    v16bf pf = load_frag(pw, 40, lane);
#pragma unroll
    for (int c = 0; c < 8; ++c) {
      v16bf vf = load_frag(&vLDS[(c * 16) * 40], 40, lane);
      O[c] = wmma_bf16(pf, vf, O[c]);
    }
  }

  int rbase = (lane >> 4) << 3;
#pragma unroll
  for (int c = 0; c < 8; ++c) {
#pragma unroll
    for (int v = 0; v < 8; ++v) {
      int m = m0 + v + rbase;
      int n = head * HD + c * 16 + (lane & 15);
      cat[(size_t)m * CATD + n] = f2bf(O[c][v] / lI[v]);
    }
  }
}

// -- kernel 8: GEMM2 out = x + gate*(cat @ lin2_w.T + b + LoRA up) ------------
__global__ void __launch_bounds__(256)
k_gemm2(const u16* __restrict__ A, const float* __restrict__ W,
        const float* __restrict__ bias, const float* __restrict__ t2,
        const float* __restrict__ up, const float* __restrict__ xin,
        const float* __restrict__ gate, float* __restrict__ out) {
  __shared__ u16 As[2 * 128 * ASTRIDE];
  __shared__ u16 Bs[2 * 256 * ASTRIDE];
  int lane = threadIdx.x & 31;
  int wave = threadIdx.x >> 5;
  int wm = (wave & 1) * 64;
  int wn = (wave >> 1) * 64;
  int m0 = blockIdx.x * 128;
  int n0 = blockIdx.y * 256;
  v8f c[4][4];
#pragma unroll
  for (int i = 0; i < 4; ++i)
#pragma unroll
    for (int j = 0; j < 4; ++j) c[i][j] = zero8();

  gemm_mainloop(A, W, CATD, m0, n0, As, Bs, c);

  int mB = m0 + wm + ((lane >> 4) << 3);
  int nB = n0 + wn + (lane & 15);
#pragma unroll
  for (int i = 0; i < 4; ++i) {
#pragma unroll
    for (int j = 0; j < 4; ++j) {
#pragma unroll
      for (int v = 0; v < 8; ++v) {
        int m = mB + i * 16 + v;
        int n = nB + j * 16;
        float val = c[i][j][v] + bias[n];
        const float* upr = up + (size_t)n * RANK;
        const float* tr = t2 + (size_t)m * RANK;
        float lo = 0.f;
#pragma unroll
        for (int r = 0; r < RANK; ++r) lo += tr[r] * upr[r];
        val += lo;
        size_t io = (size_t)m * DIM + n;
        out[io] = xin[io] + gate[n] * val;
      }
    }
  }
}

extern "C" void kernel_launch(void* const* d_in, const int* in_sizes, int n_in,
                              void* d_out, int out_size, void* d_ws, size_t ws_size,
                              hipStream_t stream) {
  (void)in_sizes; (void)n_in; (void)out_size; (void)ws_size;
  const float* x        = (const float*)d_in[0];
  const float* vec      = (const float*)d_in[1];
  const float* pe       = (const float*)d_in[2];
  const float* mod_w    = (const float*)d_in[3];
  const float* mod_b    = (const float*)d_in[4];
  const float* lin1_w   = (const float*)d_in[5];
  const float* lin1_b   = (const float*)d_in[6];
  const float* lin2_w   = (const float*)d_in[7];
  const float* lin2_b   = (const float*)d_in[8];
  const float* q_scale  = (const float*)d_in[9];
  const float* k_scale  = (const float*)d_in[10];
  const float* qkv_down = (const float*)d_in[11];
  const float* qkv_up   = (const float*)d_in[12];
  const float* proj_down= (const float*)d_in[13];
  const float* proj_up  = (const float*)d_in[14];
  float* out = (float*)d_out;

  char* base = (char*)d_ws;
  size_t off = 0;
  auto take = [&](size_t bytes) -> void* {
    void* p = base + off;
    off += bytes;
    off = (off + 255) & ~(size_t)255;
    return p;
  };
  float* mod  = (float*)take((size_t)D3 * 4);
  float* t1   = (float*)take((size_t)SEQ * RANK * 4);
  float* t2   = (float*)take((size_t)SEQ * RANK * 4);
  u16*   xmod = (u16*)  take((size_t)SEQ * DIM * 2);
  float* qkv  = (float*)take((size_t)SEQ * D3 * 4);
  u16*   cat  = (u16*)  take((size_t)SEQ * CATD * 2);
  u16*   qb   = (u16*)  take((size_t)HEADS * SEQ * HD * 2);
  u16*   kb   = (u16*)  take((size_t)HEADS * SEQ * HD * 2);
  u16*   vb   = (u16*)  take((size_t)HEADS * SEQ * HD * 2);

  k_mod<<<D3 / 8, 256, 0, stream>>>(vec, mod_w, mod_b, mod);
  k_lnmod<<<SEQ, 256, 0, stream>>>(x, mod, xmod);
  k_lora_down<<<SEQ / 8, 256, 0, stream>>>(xmod, qkv_down, t1, DIM);
  k_gemm1<<<dim3(SEQ / 128, H1 / 256), 256, 0, stream>>>(xmod, lin1_w, lin1_b, t1, qkv_up, qkv, cat);
  k_qkv_prep<<<(SEQ * HEADS) / 8, 256, 0, stream>>>(qkv, pe, q_scale, k_scale, qb, kb, vb);
  k_attn<<<dim3(SEQ / 128, HEADS), 256, 0, stream>>>(qb, kb, vb, cat);
  k_lora_down<<<SEQ / 8, 256, 0, stream>>>(cat, proj_down, t2, CATD);
  k_gemm2<<<dim3(SEQ / 128, DIM / 256), 256, 0, stream>>>(cat, lin2_w, lin2_b, t2, proj_up, x, mod + 2 * DIM, out);
}
